// KokoroModel_19378892439916
// MI455X (gfx1250) — matrix operand order repaired and checked by reading, not compiled, not run on hardware
//
#include <hip/hip_runtime.h>

// ---------------------------------------------------------------------------
// CDNA5 (gfx1250) implementation of the Kokoro-style TTS reference.
// All dense math runs through v_wmma_f32_16x16x32_bf16 (fp32 accumulate).
// Weights are pre-converted to bf16 once per launch (L2-resident, ~18 MB).
// Recurrent LSTM scans run as persistent single-workgroup kernels with
// h (bf16) and c (fp32) resident in LDS.
// ---------------------------------------------------------------------------

typedef __attribute__((ext_vector_type(16))) __bf16 v16bf;
typedef __attribute__((ext_vector_type(8)))  __bf16 v8bf;
typedef __attribute__((ext_vector_type(4)))  __bf16 v4bf;
typedef __attribute__((ext_vector_type(8)))  float  v8f;

// branch-free transcendentals on v_exp_f32 / v_rcp_f32 (no EXEC divergence,
// correct saturation at the extremes) — libm tanhf is piecewise + saveexec.
__device__ __forceinline__ float fast_rcp(float x) {
    return __builtin_amdgcn_rcpf(x);
}
__device__ __forceinline__ float sigf(float x) {
    return fast_rcp(1.0f + __expf(-x));
}
__device__ __forceinline__ float tanh_fast(float x) {
    return 1.0f - 2.0f * fast_rcp(1.0f + __expf(2.0f * x));
}

// convert a float4 into 4 consecutive bf16 lanes of a fragment
__device__ __forceinline__ void cvt4(v16bf& d, int base, float4 v) {
    d[base + 0] = (__bf16)v.x;
    d[base + 1] = (__bf16)v.y;
    d[base + 2] = (__bf16)v.z;
    d[base + 3] = (__bf16)v.w;
}

// load 16 consecutive K elements of the W operand as a bf16 fragment
__device__ __forceinline__ v16bf load_w16(const float* p) {
    const float4* pw = (const float4*)p;
    v16bf b;
    cvt4(b, 0,  pw[0]);
    cvt4(b, 4,  pw[1]);
    cvt4(b, 8,  pw[2]);
    cvt4(b, 12, pw[3]);
    return b;
}
__device__ __forceinline__ v16bf load_w16(const __bf16* p) {
    v8bf lo = *(const v8bf*)p;
    v8bf hi = *(const v8bf*)(p + 8);
    return __builtin_shufflevector(lo, hi,
        0, 1, 2, 3, 4, 5, 6, 7, 8, 9, 10, 11, 12, 13, 14, 15);
}

// ---------------------------------------------------------------------------
// Generic batched/strided GEMM.
//   WT = 1: C = scale*(A @ W^T) + bias   (W stored [N,K])
//   WT = 0: C = scale*(A @ W)   + bias   (W stored [K,N])
// WSrc = float (activations) or __bf16 (pre-converted weights).
// Batch z decomposes as outer=z/inner, in=z%inner with independent strides
// per operand (handles [B, NH] attention heads).
// One wave per 16x16 tile; M, N multiples of 16 (true throughout this model).
// K: unguarded 32-chunks + one clamped tail chunk (only the K=80 case).
// A/B: contiguous K per lane half (k = 16g + i, same in both); C layout per CDNA5 ISA 7.12.2.
// ---------------------------------------------------------------------------
template <int WT, typename WSrc>
__global__ __launch_bounds__(128)
void gemm_wmma(
    const float* __restrict__ A, int lda, long long aO, long long aI,
    const WSrc* __restrict__ W, int ldw, long long wO, long long wI,
    const float* __restrict__ bias0, const float* __restrict__ bias1,
    float* __restrict__ C, int ldc, long long cO, long long cI,
    int M, int N, int K, int innerCount, float scale)
{
    const int wave = threadIdx.x >> 5;
    const int lane = threadIdx.x & 31;
    const int g = lane >> 4;      // lane half
    const int l = lane & 15;
    const int tileN = (blockIdx.x * 4 + wave) * 16;
    const int tileM = blockIdx.y * 16;
    if (tileN >= N || tileM >= M) return;   // wave-uniform exit (EXEC stays full)

    const int z = blockIdx.z;
    const int outer = z / innerCount;
    const int inner = z - outer * innerCount;
    A += outer * aO + inner * aI;
    W += outer * wO + inner * wI;
    C += outer * cO + inner * cI;

    const int m = tileM + l;                 // A-fragment row owned by lane
    const int n = tileN + l;                 // B-fragment / C column owned by lane
    const float* __restrict__ arow = A + (long long)m * lda;
    const WSrc*  __restrict__ wrow = WT ? (W + (long long)n * ldw) : nullptr;

    v8f acc = {0.f, 0.f, 0.f, 0.f, 0.f, 0.f, 0.f, 0.f};

    const int kmain = K & ~31;
    #pragma unroll 2
    for (int k0 = 0; k0 < kmain; k0 += 32) {
        // A fragment: lane half g holds K [g*16, g*16+16)
        v16bf a;
        {
            const float4* p0 = (const float4*)(arow + k0 + g * 16);
            const float4* p1 = (const float4*)(arow + k0 + g * 16 + 8);
            cvt4(a, 0,  p0[0]);
            cvt4(a, 4,  p0[1]);
            cvt4(a, 8,  p1[0]);
            cvt4(a, 12, p1[1]);
        }
        // B fragment: lane half g holds K rows [g*16, g*16+16) of column n
        v16bf b;
        if (WT) {
            b = load_w16(wrow + k0 + g * 16);
        } else {
            #pragma unroll
            for (int i = 0; i < 16; ++i)
                b[i] = (__bf16)W[(long long)(k0 + g * 16 + i) * ldw + n];
        }
        acc = __builtin_amdgcn_wmma_f32_16x16x32_bf16(
                  false, a, false, b, (short)0, acc, false, false);
    }

    if (kmain < K) {   // clamped tail chunk (branch-free per element)
        v16bf a, b;
        #pragma unroll
        for (int i = 0; i < 16; ++i) {
            int k = kmain + g * 16 + i;
            int kc = k < K ? k : K - 1;
            float v = arow[kc];
            if (k >= K) v = 0.0f;
            a[i] = (__bf16)v;
        }
        #pragma unroll
        for (int i = 0; i < 16; ++i) {
            int k = kmain + g * 16 + i;
            int kc = k < K ? k : K - 1;
            float v = WT ? (float)wrow[kc] : (float)W[(long long)kc * ldw + n];
            if (k >= K) v = 0.0f;
            b[i] = (__bf16)v;
        }
        acc = __builtin_amdgcn_wmma_f32_16x16x32_bf16(
                  false, a, false, b, (short)0, acc, false, false);
    }

    float bsum = 0.0f;
    if (bias0) bsum += bias0[n];
    if (bias1) bsum += bias1[n];
    #pragma unroll
    for (int r = 0; r < 8; ++r) {
        int mm = tileM + g * 8 + r;          // C layout: VGPR r -> row g*8+r
        C[(long long)mm * ldc + n] = acc[r] * scale + bsum;
    }
}

// ---------------------------------------------------------------------------
// Persistent LSTM scan. One 1024-thread workgroup (32 waves) per direction.
// xg = precomputed input contribution [B, T, 4H] (+bih+bhh), B=16, H=512.
// Recurrent GEMM per step: h[16,512] x Whh^T[512,2048] via bf16 WMMA, with
// Whh pre-converted to bf16 (halves serial-path traffic, no cvt in the loop).
// Wave w owns output columns j = w*16 .. w*16+15 and computes ALL FOUR gate
// tiles (i,f,g,o) for that j-slice, so the cell update is wave-local.
// h kept bf16 in LDS (WMMA A operand, read as ds_load_b128), c fp32 in LDS.
// NOTE: K-loop is '#pragma unroll 1' on purpose — full unroll makes all Whh
// loads t-invariant, LICM preloads ~2KB/lane and spills to scratch (seen in
// round-3 disasm as "Folded Reload" scratch_load_b128 before every WMMA).
// ---------------------------------------------------------------------------
__global__ __launch_bounds__(1024)
void lstm_wmma(
    const float* __restrict__ xg0, const __bf16* __restrict__ Whh0,
    float* __restrict__ hs0, int rev0,
    const float* __restrict__ xg1, const __bf16* __restrict__ Whh1,
    float* __restrict__ hs1, int rev1,
    int T)
{
    const float* xg; const __bf16* Whh; float* hs; int rev;
    if (blockIdx.x == 0) { xg = xg0; Whh = Whh0; hs = hs0; rev = rev0; }
    else {
        if (xg1 == nullptr) return;
        xg = xg1; Whh = Whh1; hs = hs1; rev = rev1;
    }

    __shared__ float  sh_c [16 * 512];   // 32 KB: cell state fp32
    __shared__ __bf16 sh_hb[16 * 512];   // 16 KB: hidden state bf16 (WMMA A)

    const int tid  = threadIdx.x;
    const int wave = tid >> 5;
    const int lane = tid & 31;
    const int g = lane >> 4;
    const int l = lane & 15;
    const int jbase = wave * 16;

    for (int i = tid; i < 16 * 512; i += 1024) {
        sh_c[i]  = 0.0f;
        sh_hb[i] = (__bf16)0.0f;
    }
    __syncthreads();

    // per-wave invariant row bases for Whh (gate e, column slice jbase+l)
    const __bf16* wbase[4];
    #pragma unroll
    for (int e = 0; e < 4; ++e)
        wbase[e] = Whh + (long long)(e * 512 + jbase + l) * 512;

    for (int t = 0; t < T; ++t) {
        const int tc = rev ? (T - 1 - t) : t;

        // init accumulators from xg (gate order i,f,g,o)
        v8f acc[4];
        #pragma unroll
        for (int e = 0; e < 4; ++e) {
            #pragma unroll
            for (int r = 0; r < 8; ++r) {
                int b = g * 8 + r;
                acc[e][r] = xg[((long long)(b * T + tc)) * 2048
                               + e * 512 + jbase + l];
            }
        }

        // recurrent GEMM: K = 512 in 16 WMMA steps, A frag shared by 4 gates
        #pragma unroll 1
        for (int k0 = 0; k0 < 512; k0 += 32) {
            v8bf lo = *(const v8bf*)(&sh_hb[l * 512 + k0 + g * 16]);
            v8bf hi = *(const v8bf*)(&sh_hb[l * 512 + k0 + g * 16 + 8]);
            v16bf a = __builtin_shufflevector(lo, hi,
                        0, 1, 2, 3, 4, 5, 6, 7, 8, 9, 10, 11, 12, 13, 14, 15);
            #pragma unroll
            for (int e = 0; e < 4; ++e) {
                v16bf b = load_w16(wbase[e] + k0 + g * 16);
                acc[e] = __builtin_amdgcn_wmma_f32_16x16x32_bf16(
                             false, a, false, b, (short)0, acc[e], false, false);
            }
        }

        __syncthreads();   // everyone done reading sh_hb before we rewrite it

        // cell update: each (lane, r) owns a unique (batch b, column j)
        #pragma unroll
        for (int r = 0; r < 8; ++r) {
            int b = g * 8 + r;
            int j = jbase + l;
            float ig = sigf(acc[0][r]);
            float fg = sigf(acc[1][r]);
            float gg = tanh_fast(acc[2][r]);
            float og = sigf(acc[3][r]);
            float c  = fg * sh_c[b * 512 + j] + ig * gg;
            float h  = og * tanh_fast(c);
            sh_c[b * 512 + j]  = c;
            sh_hb[b * 512 + j] = (__bf16)h;
            hs[((long long)(b * T + tc)) * 512 + j] = h;   // [B,T,H]
        }
        __syncthreads();
    }
}

// ---------------------------------------------------------------------------
// Small helper kernels
// ---------------------------------------------------------------------------
__global__ void bf16cvt_k(const float* __restrict__ src, __bf16* __restrict__ dst,
                          long long n4)
{
    long long i = (long long)blockIdx.x * 256 + threadIdx.x;
    if (i >= n4) return;
    float4 v = ((const float4*)src)[i];
    v4bf o;
    o[0] = (__bf16)v.x; o[1] = (__bf16)v.y;
    o[2] = (__bf16)v.z; o[3] = (__bf16)v.w;
    ((v4bf*)dst)[i] = o;
}

__global__ void embed_k(const int* __restrict__ idx, const float* __restrict__ emb,
                        float* __restrict__ x)
{
    int row = blockIdx.x;           // b*Ttxt + t
    int t = threadIdx.x;            // 128 threads; 128 float4 = 512 floats
    int e = idx[row];
    ((float4*)x)[(long long)row * 128 + t] =
        ((const float4*)emb)[(long long)e * 128 + t];
}

__global__ void melshift_k(const float* __restrict__ mel, float* __restrict__ out,
                           int T, int M)
{
    int row = blockIdx.x;           // b*T + t
    int m = threadIdx.x;            // M threads
    int t = row % T;
    out[(long long)row * M + m] =
        (t == 0) ? 0.0f : mel[(long long)(row - 1) * M + m];
}

// concat two [rows, 512] into [rows, 1024]; one row per block, 256 float4 lanes
__global__ void concat2_k(const float* __restrict__ a, const float* __restrict__ b,
                          float* __restrict__ o)
{
    long long row = blockIdx.x;
    int t = threadIdx.x;            // 256 threads
    float4 v = (t < 128) ? ((const float4*)a)[row * 128 + t]
                         : ((const float4*)b)[row * 128 + (t - 128)];
    ((float4*)o)[row * 256 + t] = v;
}

__global__ void softmax256_k(float* __restrict__ s)
{
    __shared__ float red[256];
    long long row = blockIdx.x;
    float* p = s + row * 256;
    int t = threadIdx.x;
    float v = p[t];
    red[t] = v; __syncthreads();
    for (int o = 128; o > 0; o >>= 1) {
        if (t < o) red[t] = fmaxf(red[t], red[t + o]);
        __syncthreads();
    }
    float mx = red[0]; __syncthreads();
    float e = __expf(v - mx);
    red[t] = e; __syncthreads();
    for (int o = 128; o > 0; o >>= 1) {
        if (t < o) red[t] += red[t + o];
        __syncthreads();
    }
    p[t] = e * fast_rcp(red[0]);
}

// ---------------------------------------------------------------------------
// Host-side orchestration (graph-capture safe: launches only, all on stream)
// ---------------------------------------------------------------------------
extern "C" void kernel_launch(void* const* d_in, const int* in_sizes, int n_in,
                              void* d_out, int out_size, void* d_ws, size_t ws_size,
                              hipStream_t stream)
{
    (void)n_in; (void)out_size; (void)ws_size;
    const int B = 16, H = 512, MEL = 80, NHh = 8;
    const int Ttxt = in_sizes[0] / B;              // 256
    const int Tmel = in_sizes[1] / (B * MEL);      // 400

    const int*   idx    = (const int*)  d_in[0];
    const float* mel    = (const float*)d_in[1];
    const float* emb    = (const float*)d_in[2];
    const float* efWih  = (const float*)d_in[3];
    const float* efWhh  = (const float*)d_in[4];
    const float* efBih  = (const float*)d_in[5];
    const float* efBhh  = (const float*)d_in[6];
    const float* ebWih  = (const float*)d_in[7];
    const float* ebWhh  = (const float*)d_in[8];
    const float* ebBih  = (const float*)d_in[9];
    const float* ebBhh  = (const float*)d_in[10];
    const float* tpW    = (const float*)d_in[11];
    const float* tpB    = (const float*)d_in[12];
    const float* mpiW   = (const float*)d_in[13];
    const float* mpiB   = (const float*)d_in[14];
    const float* ainW   = (const float*)d_in[15];
    const float* ainB   = (const float*)d_in[16];
    const float* aoutW  = (const float*)d_in[17];
    const float* aoutB  = (const float*)d_in[18];
    const float* decWih = (const float*)d_in[19];
    const float* decWhh = (const float*)d_in[20];
    const float* decBih = (const float*)d_in[21];
    const float* decBhh = (const float*)d_in[22];
    const float* mpoW   = (const float*)d_in[23];
    const float* mpoB   = (const float*)d_in[24];
    float* out = (float*)d_out;

    // ---- workspace arenas (fp32 elements), lifetime-aliased ----
    float* ws = (float*)d_ws;
    size_t off = 0;
    auto alloc = [&](size_t nel) { float* p = ws + off; off += nel; return p; };

    const size_t rT = (size_t)B * Ttxt;            // 4096 rows (text)
    const size_t rM = (size_t)B * Tmel;            // 6400 rows (mel)

    float* arenaA = alloc(2 * rT * 2048);          // xgf|xgb -> scores -> xgd
    float* arenaB = alloc(rT * H * 2 + rT * 2 * H);// hf|hb|cat -> q|k|v -> ao|att
    float* xbuf   = alloc(rT * H);                 // embedded text -> mel_shifted
    float* te     = alloc(rT * H);                 // text_encoded
    float* mp     = alloc(rM * H);                 // mel_projected
    float* di     = alloc(rM * 2 * H);             // decoder input
    float* hd     = alloc(rM * H);                 // decoder hidden seq

    // bf16 weight arena (allocated in fp32 units, used as __bf16)
    size_t bfoff = 0;
    __bf16* bfbase = (__bf16*)alloc(4600000);      // 9.2M bf16 capacity
    auto balloc = [&](size_t nel) { __bf16* p = bfbase + bfoff; bfoff += nel; return p; };
    __bf16* b_efWih  = balloc((size_t)2048 * 512);
    __bf16* b_ebWih  = balloc((size_t)2048 * 512);
    __bf16* b_efWhh  = balloc((size_t)2048 * 512);
    __bf16* b_ebWhh  = balloc((size_t)2048 * 512);
    __bf16* b_tpW    = balloc((size_t)512 * 1024);
    __bf16* b_mpiW   = balloc((size_t)512 * 80);
    __bf16* b_ainW   = balloc((size_t)1536 * 512);
    __bf16* b_aoutW  = balloc((size_t)512 * 512);
    __bf16* b_decWih = balloc((size_t)2048 * 1024);
    __bf16* b_decWhh = balloc((size_t)2048 * 512);
    __bf16* b_mpoW   = balloc((size_t)80 * 512);

    float* xgf = arenaA;
    float* xgb = arenaA + rT * 2048;
    float* scores = arenaA;                        // B*NH*Tmel*Ttxt fp32
    float* xgd = arenaA;                           // rM*2048 fp32
    float* hf  = arenaB;
    float* hb  = arenaB + rT * H;
    float* cat = arenaB + 2 * rT * H;
    float* qb  = arenaB;                           // rM*H
    float* kb  = arenaB + rM * H;                  // rT*H
    float* vb  = kb + rT * H;                      // rT*H
    float* ao  = arenaB;                           // rM*H (q dead)
    float* att = arenaB + rM * H;                  // rM*H (k,v dead)
    float* melsh = xbuf;                           // rM*MEL (x dead)

    auto cvt = [&](const float* src, __bf16* dst, size_t n) {
        long long n4 = (long long)(n / 4);
        bf16cvt_k<<<dim3((unsigned)((n4 + 255) / 256)), dim3(256), 0, stream>>>(
            src, dst, n4);
    };
    // bf16 GEMM against pre-converted weights (W stored [N,K])
    auto gemm_bw = [&](const float* A, int lda,
                       const __bf16* W, int ldw,
                       const float* b0, const float* b1,
                       float* C, int ldc, int Mm, int Nn, int Kk) {
        dim3 grid((Nn + 63) / 64, (Mm + 15) / 16, 1);
        gemm_wmma<1, __bf16><<<grid, dim3(128), 0, stream>>>(
            A, lda, 0, 0, W, ldw, 0, 0, b0, b1,
            C, ldc, 0, 0, Mm, Nn, Kk, 1, 1.0f);
    };

    // 0) pre-convert all weight matrices to bf16 (once per launch)
    cvt(efWih,  b_efWih,  (size_t)2048 * 512);
    cvt(ebWih,  b_ebWih,  (size_t)2048 * 512);
    cvt(efWhh,  b_efWhh,  (size_t)2048 * 512);
    cvt(ebWhh,  b_ebWhh,  (size_t)2048 * 512);
    cvt(tpW,    b_tpW,    (size_t)512 * 1024);
    cvt(mpiW,   b_mpiW,   (size_t)512 * 80);
    cvt(ainW,   b_ainW,   (size_t)1536 * 512);
    cvt(aoutW,  b_aoutW,  (size_t)512 * 512);
    cvt(decWih, b_decWih, (size_t)2048 * 1024);
    cvt(decWhh, b_decWhh, (size_t)2048 * 512);
    cvt(mpoW,   b_mpoW,   (size_t)80 * 512);

    // 1) embedding gather -> x [B*Ttxt, 512]
    embed_k<<<dim3((unsigned)rT), dim3(128), 0, stream>>>(idx, emb, xbuf);

    // 2,3) encoder input contributions xg = x @ Wih^T + bih + bhh  [B,T,4H]
    gemm_bw(xbuf, H, b_efWih, H, efBih, efBhh, xgf, 2048, (int)rT, 2048, H);
    gemm_bw(xbuf, H, b_ebWih, H, ebBih, ebBhh, xgb, 2048, (int)rT, 2048, H);

    // 4) fwd + bwd encoder LSTM scans (two persistent workgroups)
    lstm_wmma<<<dim3(2), dim3(1024), 0, stream>>>(
        xgf, b_efWhh, hf, 0, xgb, b_ebWhh, hb, 1, Ttxt);

    // 5) concat [h_f, h_b] -> cat [B*Ttxt, 1024]
    concat2_k<<<dim3((unsigned)rT), dim3(256), 0, stream>>>(hf, hb, cat);

    // 6) text projection: te = cat @ tpW^T + tpB
    gemm_bw(cat, 2 * H, b_tpW, 2 * H, tpB, nullptr, te, H, (int)rT, H, 2 * H);

    // 7) teacher-forcing shift of mel
    melshift_k<<<dim3((unsigned)rM), dim3(MEL), 0, stream>>>(mel, melsh, Tmel, MEL);

    // 8) mel input projection: mp = melsh @ mpiW^T + mpiB   (K=80 tail path)
    gemm_bw(melsh, MEL, b_mpiW, MEL, mpiB, nullptr, mp, H, (int)rM, H, MEL);

    // 9-11) q/k/v projections
    gemm_bw(mp, H, b_ainW, H, ainB, nullptr, qb, H, (int)rM, H, H);
    gemm_bw(te, H, b_ainW + (size_t)H * H, H, ainB + H, nullptr,
            kb, H, (int)rT, H, H);
    gemm_bw(te, H, b_ainW + (size_t)2 * H * H, H, ainB + 2 * H, nullptr,
            vb, H, (int)rT, H, H);

    // 12) scores[b,h,q,k] = (Q . K^T) / 8 : batched over B*NH heads
    {
        dim3 grid((Ttxt + 63) / 64, (Tmel + 15) / 16, B * NHh);
        gemm_wmma<1, float><<<grid, dim3(128), 0, stream>>>(
            qb, H, (long long)Tmel * H, 64,
            kb, H, (long long)Ttxt * H, 64, nullptr, nullptr,
            scores, Ttxt, (long long)NHh * Tmel * Ttxt, (long long)Tmel * Ttxt,
            Tmel, Ttxt, 64, NHh, 0.125f);
    }

    // 13) softmax over keys (rows of 256)
    softmax256_k<<<dim3((unsigned)(B * NHh * Tmel)), dim3(256), 0, stream>>>(scores);

    // 14) attn @ V -> ao [B, Tmel, NH*HD]  (W non-transposed path)
    {
        dim3 grid((64 + 63) / 64, (Tmel + 15) / 16, B * NHh);
        gemm_wmma<0, float><<<grid, dim3(128), 0, stream>>>(
            scores, Ttxt, (long long)NHh * Tmel * Ttxt, (long long)Tmel * Ttxt,
            vb, H, (long long)Ttxt * H, 64, nullptr, nullptr,
            ao, H, (long long)Tmel * H, 64,
            Tmel, 64, Ttxt, NHh, 1.0f);
    }

    // 15) attention output projection
    gemm_bw(ao, H, b_aoutW, H, aoutB, nullptr, att, H, (int)rM, H, H);

    // 16) decoder input = concat(mp, att) [B*Tmel, 1024]
    concat2_k<<<dim3((unsigned)rM), dim3(256), 0, stream>>>(mp, att, di);

    // 17) decoder input contribution: xgd = di @ decWih^T + bih + bhh (K=1024)
    gemm_bw(di, 2 * H, b_decWih, 2 * H, decBih, decBhh,
            xgd, 2048, (int)rM, 2048, 2 * H);

    // 18) decoder LSTM scan
    lstm_wmma<<<dim3(1), dim3(1024), 0, stream>>>(
        xgd, b_decWhh, hd, 0, nullptr, nullptr, nullptr, 0, Tmel);

    // 19) mel output projection -> d_out [B, Tmel, 80]
    gemm_bw(hd, H, b_mpoW, H, mpoB, nullptr, out, MEL, (int)rM, MEL, H);
}
